// SALAD_27298812133797
// MI455X (gfx1250) — matrix operand order repaired and checked by reading, not compiled, not run on hardware
//
#include <hip/hip_runtime.h>
#include <hip/hip_bf16.h>

typedef unsigned short u16;
typedef __attribute__((ext_vector_type(16))) __bf16 v16bf;
typedef __attribute__((ext_vector_type(8)))  float  v8f;

struct Frag32 { uint4 lo, hi; };   // 32 bytes == v16bf

__device__ __forceinline__ u16 f2bf(float f) {
    unsigned int u = __float_as_uint(f);
    unsigned int r = (u + 0x7FFFu + ((u >> 16) & 1u)) >> 16;
    return (u16)r;
}

// ---------------------------------------------------------------------------
// bf16 WMMA GEMM, MTxNT block tile, double-buffered LDS, register-pipelined
// global->LDS staging. Each wave owns a WMx64 strip (WM/16 x 4 accumulators).
// BF32: B matrix is fp32 in global memory, converted to bf16 during staging.
// Per-thread staging is exactly 2 x b128 for A and 2 chunks for B in both
// tile configurations (THREADS = 2*NT, A tile = MT*4 uint4, B tile = NT*4).
// ---------------------------------------------------------------------------
template <int MT, int NT, int WM, bool BF32>
__global__ __launch_bounds__((MT / WM) * (NT / 64) * 32)
void gemm_wmma(const u16* __restrict__ A, long long strideA, int lda,
               const void* __restrict__ Bv, long long strideB, int ldb,
               const float* __restrict__ bias, int strideBias,
               float* __restrict__ Cf, u16* __restrict__ Cb,
               long long strideC, int ldc, int Kd, int relu)
{
    constexpr int WAVES_M = MT / WM;
    constexpr int WAVES_N = NT / 64;
    constexpr int THREADS = WAVES_M * WAVES_N * 32;
    constexpr int FM      = WM / 16;     // A fragments per wave
    constexpr int GB      = NT / 8;      // 8-elem groups per B row

    __shared__ u16 As[2][MT][32];
    __shared__ u16 Bs[2][NT][32];        // transposed: Bs[buf][n][k]

    const int bz   = blockIdx.z;
    const int m0   = blockIdx.y * MT;
    const int n0   = blockIdx.x * NT;
    const int tid  = threadIdx.x;
    const int wave = tid >> 5;
    const int lane = tid & 31;
    const int lm   = lane & 15;
    const int lh   = lane >> 4;
    const int wm   = wave % WAVES_M;
    const int wn   = wave / WAVES_M;

    const u16*   Ab = A + (long long)bz * strideA;
    const u16*   Bh = ((const u16*)Bv)   + (long long)bz * strideB;
    const float* Bf = ((const float*)Bv) + (long long)bz * strideB;

    // ---- per-thread staging descriptors (2 chunks each for A and B) ----
    int ar[2], aq[2], bk[2], bg[2];
    const u16*   aptr[2];
    const u16*   bhptr[2];
    const float* bfptr[2];
#pragma unroll
    for (int i = 0; i < 2; ++i) {
        const int fl = tid + i * THREADS;
        ar[i] = fl >> 2;  aq[i] = fl & 3;       // A: row, quarter (32/8 = 4 per row)
        bk[i] = fl / GB;  bg[i] = fl % GB;      // B: k-row, 8-col group
        aptr[i]  = Ab + (long long)(m0 + ar[i]) * lda + aq[i] * 8;
        bhptr[i] = Bh + (long long)bk[i] * ldb + n0 + bg[i] * 8;
        bfptr[i] = Bf + (long long)bk[i] * ldb + n0 + bg[i] * 8;
    }

    uint4  rA[2];
    uint4  rB[2];
    float4 rBf[2][2];

    auto load_regs = [&](int k0) {
#pragma unroll
        for (int i = 0; i < 2; ++i) {
            rA[i] = *reinterpret_cast<const uint4*>(aptr[i] + k0);
            if (BF32) {
                const float* s = bfptr[i] + (long long)k0 * ldb;
                rBf[i][0] = *reinterpret_cast<const float4*>(s);
                rBf[i][1] = *reinterpret_cast<const float4*>(s + 4);
            } else {
                rB[i] = *reinterpret_cast<const uint4*>(bhptr[i] + (long long)k0 * ldb);
            }
        }
    };
    auto store_lds = [&](int buf) {
#pragma unroll
        for (int i = 0; i < 2; ++i) {
            *reinterpret_cast<uint4*>(&As[buf][ar[i]][aq[i] * 8]) = rA[i];
            if (BF32) {
                const float f[8] = {rBf[i][0].x, rBf[i][0].y, rBf[i][0].z, rBf[i][0].w,
                                    rBf[i][1].x, rBf[i][1].y, rBf[i][1].z, rBf[i][1].w};
#pragma unroll
                for (int j = 0; j < 8; ++j) Bs[buf][bg[i] * 8 + j][bk[i]] = f2bf(f[j]);
            } else {
                const u16* ds = reinterpret_cast<const u16*>(&rB[i]);
#pragma unroll
                for (int j = 0; j < 8; ++j) Bs[buf][bg[i] * 8 + j][bk[i]] = ds[j];
            }
        }
    };

    v8f acc[FM][4] = {};

    load_regs(0);
    store_lds(0);
    __syncthreads();

    int cur = 0;
    for (int k0 = 0; k0 < Kd; k0 += 32) {
        const bool more = (k0 + 32) < Kd;
        if (more) load_regs(k0 + 32);                 // in-flight during WMMAs
        if (k0 + 64 < Kd) {                           // tile after next -> L2
            __builtin_prefetch(aptr[0] + k0 + 64, 0, 0);
            if (BF32) __builtin_prefetch(bfptr[0] + (long long)(k0 + 64) * ldb, 0, 0);
            else      __builtin_prefetch(bhptr[0] + (long long)(k0 + 64) * ldb, 0, 0);
        }

        // A fragments: lane(m, half=lh) -> K[8lh..8lh+7] , K[16+8lh..16+8lh+7]
        v16bf av[FM];
#pragma unroll
        for (int fi = 0; fi < FM; ++fi) {
            const int am = wm * WM + fi * 16 + lm;
            Frag32 fa;
            fa.lo = *reinterpret_cast<const uint4*>(&As[cur][am][lh * 8]);
            fa.hi = *reinterpret_cast<const uint4*>(&As[cur][am][16 + lh * 8]);
            av[fi] = __builtin_bit_cast(v16bf, fa);
        }
#pragma unroll
        for (int t = 0; t < 4; ++t) {
            const int bn = wn * 64 + t * 16 + lm;
            Frag32 fb;
            fb.lo = *reinterpret_cast<const uint4*>(&Bs[cur][bn][lh * 8]);
            fb.hi = *reinterpret_cast<const uint4*>(&Bs[cur][bn][16 + lh * 8]);
            v16bf bv = __builtin_bit_cast(v16bf, fb);
#pragma unroll
            for (int fi = 0; fi < FM; ++fi)
                acc[fi][t] = __builtin_amdgcn_wmma_f32_16x16x32_bf16(
                    false, av[fi], false, bv, (short)0, acc[fi][t], false, false);
        }

        if (more) store_lds(cur ^ 1);                 // waits loads, fills other buffer
        __syncthreads();
        cur ^= 1;
    }

    // ---- epilogue (C layout: n = lane%16, m = r + 8*(lane/16)) ----
    const float* biasb = bias ? bias + (long long)bz * strideBias : nullptr;
#pragma unroll
    for (int fi = 0; fi < FM; ++fi)
#pragma unroll
    for (int t = 0; t < 4; ++t) {
        const int n = n0 + wn * 64 + t * 16 + lm;
#pragma unroll
        for (int r = 0; r < 8; ++r) {
            const int m = m0 + wm * WM + fi * 16 + r + 8 * lh;
            float v = acc[fi][t][r];
            if (biasb) v += biasb[m];
            if (relu)  v = fmaxf(v, 0.0f);
            const long long idx = (long long)bz * strideC + (long long)m * ldc + n;
            if (Cf) Cf[idx] = v;
            if (Cb) Cb[idx] = f2bf(v);
        }
    }
}

// ---------------------------------------------------------------------------
// Weight prep: Wcat = [cf_w1 ; ex_w1[0] ; ex_w1[1]]  (1536x1536) + bias concat
// ---------------------------------------------------------------------------
__global__ void build_wcat(const float* __restrict__ cf_w1,
                           const float* __restrict__ ex_w1,
                           const float* __restrict__ cf_b1,
                           const float* __restrict__ ex_b1,
                           u16* __restrict__ Wcat, float* __restrict__ bcat)
{
    int i = blockIdx.x * 256 + threadIdx.x;
    if (i >= 1536 * 1536) return;
    int r = i / 1536, c = i % 1536;
    float w = (r < 512) ? cf_w1[r * 1536 + c] : ex_w1[(r - 512) * 1536 + c];
    Wcat[i] = f2bf(w);
    if (c == 0) bcat[r] = (r < 512) ? cf_b1[r] : ex_b1[r - 512];
}

__global__ void f32_to_bf16(const float* __restrict__ in, u16* __restrict__ out, int n)
{
    int i = blockIdx.x * 256 + threadIdx.x;
    if (i < n) out[i] = f2bf(in[i]);
}

// ---------------------------------------------------------------------------
// Gate + token MLP (tiny): gw[b][2] softmax, t[b][256]
// ---------------------------------------------------------------------------
__global__ __launch_bounds__(256)
void gate_token_kernel(const float* __restrict__ token,
                       const float* __restrict__ g_w1, const float* __restrict__ g_b1,
                       const float* __restrict__ g_w2, const float* __restrict__ g_b2,
                       const float* __restrict__ tf_w1, const float* __restrict__ tf_b1,
                       const float* __restrict__ tf_w2, const float* __restrict__ tf_b2,
                       float* __restrict__ gw, float* __restrict__ tout)
{
    __shared__ float gh[128];
    __shared__ float th[512];
    __shared__ float g2[2];
    const int b = blockIdx.x, tid = threadIdx.x;
    const float* tok = token + b * 1536;

    if (tid < 128) {
        float s = g_b1[tid];
        const float* w = g_w1 + tid * 1536;
        for (int c = 0; c < 1536; ++c) s += tok[c] * w[c];
        gh[tid] = fmaxf(s, 0.f);
    }
    for (int o = tid; o < 512; o += 256) {
        float s = tf_b1[o];
        const float* w = tf_w1 + o * 1536;
        for (int c = 0; c < 1536; ++c) s += tok[c] * w[c];
        th[o] = fmaxf(s, 0.f);
    }
    __syncthreads();
    if (tid < 2) {
        float s = g_b2[tid];
        const float* w = g_w2 + tid * 128;
        for (int c = 0; c < 128; ++c) s += gh[c] * w[c];
        g2[tid] = s;
    }
    __syncthreads();
    if (tid == 0) {
        float mx = fmaxf(g2[0], g2[1]);
        float e0 = __expf(g2[0] - mx), e1 = __expf(g2[1] - mx);
        float inv = 1.f / (e0 + e1);
        gw[b * 2 + 0] = e0 * inv;
        gw[b * 2 + 1] = e1 * inv;
    }
    {   // token features
        float s = tf_b2[tid];
        const float* w = tf_w2 + tid * 512;
        for (int c = 0; c < 512; ++c) s += th[c] * w[c];
        tout[b * 256 + tid] = s;
    }
}

// ---------------------------------------------------------------------------
// Per-batch MoE-mixed weight: Wp[b][k][0:512]=gw0*W2[0][k], [512:1024]=gw1*W2[1][k]
// ---------------------------------------------------------------------------
__global__ void build_wp(const float* __restrict__ ex_w2, const float* __restrict__ ex_b2,
                         const float* __restrict__ gw,
                         u16* __restrict__ Wp, float* __restrict__ pbias)
{
    int idx = blockIdx.x * 256 + threadIdx.x;      // B*64*1024
    int b = idx >> 16, rem = idx & 65535;
    int k = rem >> 10, j = rem & 1023;
    int e = j >> 9, jj = j & 511;
    float g = gw[b * 2 + e];
    Wp[idx] = f2bf(g * ex_w2[(long long)e * 64 * 512 + k * 512 + jj]);
    if (j == 0)
        pbias[b * 64 + k] = gw[b * 2] * ex_b2[k] + gw[b * 2 + 1] * ex_b2[64 + k];
}

// ---------------------------------------------------------------------------
// Sinkhorn (log-domain), m=64, n=1024, bins row = alpha. Z is implicit (p + bin).
// ---------------------------------------------------------------------------
__global__ void zero_f32(float* __restrict__ p, int n)
{
    int i = blockIdx.x * 256 + threadIdx.x;
    if (i < n) p[i] = 0.f;
}

__global__ __launch_bounds__(256)
void sink_u(const float* __restrict__ p, const float* __restrict__ alpha,
            const float* __restrict__ v, float* __restrict__ u)
{
    const int i = blockIdx.x;   // 0..64
    const int b = blockIdx.y;
    const int tid = threadIdx.x;
    __shared__ float red[256];
    const float* vb  = v + b * 1024;
    const float* row = p + (long long)b * 64 * 1024 + (long long)i * 1024;
    const float a = *alpha;

    float mx = -1e30f;
    for (int n = tid; n < 1024; n += 256) {
        float z = a;
        if (i < 64) z = row[n];
        mx = fmaxf(mx, z + vb[n]);
    }
    red[tid] = mx; __syncthreads();
    for (int s = 128; s > 0; s >>= 1) { if (tid < s) red[tid] = fmaxf(red[tid], red[tid + s]); __syncthreads(); }
    mx = red[0]; __syncthreads();

    float sum = 0.f;
    for (int n = tid; n < 1024; n += 256) {
        float z = a;
        if (i < 64) z = row[n];
        sum += __expf(z + vb[n] - mx);
    }
    red[tid] = sum; __syncthreads();
    for (int s = 128; s > 0; s >>= 1) { if (tid < s) red[tid] += red[tid + s]; __syncthreads(); }
    if (tid == 0) {
        float lse  = mx + __logf(red[0]);
        float norm = -logf(1088.0f);                       // -(log(64+1024))
        float lmu  = (i < 64) ? norm : (logf(1024.0f) + norm);
        u[b * 65 + i] = lmu - lse;
    }
}

__global__ __launch_bounds__(1024)
void sink_v(const float* __restrict__ p, const float* __restrict__ alpha,
            const float* __restrict__ u, float* __restrict__ v)
{
    const int b = blockIdx.x;
    const int n = threadIdx.x;   // 0..1023
    __shared__ float us[65];
    if (n < 65) us[n] = u[b * 65 + n];
    __syncthreads();
    const float* pb = p + (long long)b * 64 * 1024;
    const float a = *alpha + us[64];
    float mx = a;
    for (int i = 0; i < 64; ++i) mx = fmaxf(mx, pb[i * 1024 + n] + us[i]);
    float sum = __expf(a - mx);
    for (int i = 0; i < 64; ++i) sum += __expf(pb[i * 1024 + n] + us[i] - mx);
    v[b * 1024 + n] = -logf(1088.0f) - (mx + __logf(sum));
}

// Pt[b][n][k] = bf16(exp(p[k][n] + u[k] + v[n] + log(1088)))  (transposed P)
__global__ void pexp_kernel(const float* __restrict__ p, const float* __restrict__ u,
                            const float* __restrict__ v, u16* __restrict__ Pt)
{
    int idx = blockIdx.x * 256 + threadIdx.x;      // B*1024*64
    int b = idx >> 16, rem = idx & 65535;
    int n = rem >> 6, k = rem & 63;
    float z = p[(long long)b * 65536 + k * 1024 + n] + u[b * 65 + k] + v[b * 1024 + n]
            + logf(1088.0f);
    Pt[idx] = f2bf(__expf(z));
}

// ---------------------------------------------------------------------------
// Final L2 norms + concat: out[b] = l2n([ l2n(t) , l2n(agg, over d) ])
// ---------------------------------------------------------------------------
__global__ __launch_bounds__(256)
void finalize_kernel(const float* __restrict__ t, const float* __restrict__ agg,
                     float* __restrict__ out)
{
    const int b = blockIdx.x, tid = threadIdx.x;
    __shared__ float red[256];
    __shared__ float coln[64];
    __shared__ float tns;
    const float* tb = t + b * 256;
    const float* ab = agg + (long long)b * 128 * 64;

    float x = tb[tid];
    red[tid] = x * x; __syncthreads();
    for (int s = 128; s > 0; s >>= 1) { if (tid < s) red[tid] += red[tid + s]; __syncthreads(); }
    if (tid == 0) tns = fmaxf(sqrtf(red[0]), 1e-12f);
    if (tid < 64) {
        float s = 0.f;
        for (int d = 0; d < 128; ++d) { float v = ab[d * 64 + tid]; s += v * v; }
        coln[tid] = fmaxf(sqrtf(s), 1e-12f);
    }
    __syncthreads();
    const float tn = tns;

    float s = (x / tn) * (x / tn);
    for (int i = tid; i < 8192; i += 256) {
        float v = ab[i] / coln[i & 63];
        s += v * v;
    }
    red[tid] = s; __syncthreads();
    for (int s2 = 128; s2 > 0; s2 >>= 1) { if (tid < s2) red[tid] += red[tid + s2]; __syncthreads(); }
    const float fn = fmaxf(sqrtf(red[0]), 1e-12f);

    out[b * 8448 + tid] = (x / tn) / fn;
    for (int i = tid; i < 8192; i += 256)
        out[b * 8448 + 256 + i] = (ab[i] / coln[i & 63]) / fn;   // i = d*64 + k
}

// ---------------------------------------------------------------------------
extern "C" void kernel_launch(void* const* d_in, const int* in_sizes, int n_in,
                              void* d_out, int out_size, void* d_ws, size_t ws_size,
                              hipStream_t stream)
{
    (void)in_sizes; (void)n_in; (void)out_size; (void)ws_size;
    const float* x_feat = (const float*)d_in[0];
    const float* token  = (const float*)d_in[1];
    const float* cf_w1  = (const float*)d_in[2];
    const float* cf_b1  = (const float*)d_in[3];
    const float* cf_w2  = (const float*)d_in[4];
    const float* cf_b2  = (const float*)d_in[5];
    const float* tf_w1  = (const float*)d_in[6];
    const float* tf_b1  = (const float*)d_in[7];
    const float* tf_w2  = (const float*)d_in[8];
    const float* tf_b2  = (const float*)d_in[9];
    const float* ex_w1  = (const float*)d_in[10];
    const float* ex_b1  = (const float*)d_in[11];
    const float* ex_w2  = (const float*)d_in[12];
    const float* ex_b2  = (const float*)d_in[13];
    const float* g_w1   = (const float*)d_in[14];
    const float* g_b1   = (const float*)d_in[15];
    const float* g_w2   = (const float*)d_in[16];
    const float* g_b2   = (const float*)d_in[17];
    const float* alpha  = (const float*)d_in[18];
    float* out = (float*)d_out;

    char* ws = (char*)d_ws;
    size_t off = 0;
    auto alloc = [&](size_t bytes) -> void* {
        void* p = ws + off; off += (bytes + 255) & ~(size_t)255; return p;
    };
    u16*   Wcat  = (u16*)  alloc((size_t)1536 * 1536 * 2);        // 4.7 MB
    float* bcat  = (float*)alloc(1536 * 4);
    u16*   Hcat  = (u16*)  alloc((size_t)16 * 1536 * 1024 * 2);   // 50.3 MB
    u16*   W2cf  = (u16*)  alloc((size_t)128 * 512 * 2);
    u16*   Fb    = (u16*)  alloc((size_t)16 * 128 * 1024 * 2);    // 4.2 MB
    u16*   Wp    = (u16*)  alloc((size_t)16 * 64 * 1024 * 2);     // 2.1 MB
    float* pbias = (float*)alloc(16 * 64 * 4);
    float* pbuf  = (float*)alloc((size_t)16 * 64 * 1024 * 4);     // 4.2 MB
    float* ubuf  = (float*)alloc(16 * 65 * 4);
    float* vbuf  = (float*)alloc(16 * 1024 * 4);
    u16*   Pt    = (u16*)  alloc((size_t)16 * 1024 * 64 * 2);     // 2.1 MB
    float* aggb  = (float*)alloc((size_t)16 * 128 * 64 * 4);
    float* tbuf  = (float*)alloc(16 * 256 * 4);
    float* gwb   = (float*)alloc(16 * 2 * 4);

    // weight prep
    build_wcat<<<(1536 * 1536 + 255) / 256, 256, 0, stream>>>(cf_w1, ex_w1, cf_b1, ex_b1, Wcat, bcat);
    f32_to_bf16<<<(128 * 512 + 255) / 256, 256, 0, stream>>>(cf_w2, W2cf, 128 * 512);
    // gate + token MLP, then MoE-mixed weight
    gate_token_kernel<<<16, 256, 0, stream>>>(token, g_w1, g_b1, g_w2, g_b2,
                                              tf_w1, tf_b1, tf_w2, tf_b2, gwb, tbuf);
    build_wp<<<(16 * 64 * 1024) / 256, 256, 0, stream>>>(ex_w2, ex_b2, gwb, Wp, pbias);
    zero_f32<<<(16 * 1024 + 255) / 256, 256, 0, stream>>>(vbuf, 16 * 1024);

    // Stage 1: Hcat[b] = ReLU(Wcat * X[b] + bcat)   [1536 x 1024], K=1536, B fp32
    gemm_wmma<128, 128, 32, true><<<dim3(8, 12, 16), 256, 0, stream>>>(
        Wcat, 0LL, 1536, x_feat, (long long)1536 * 1024, 1024,
        bcat, 0, nullptr, Hcat, (long long)1536 * 1024, 1024, 1536, 1);
    // Stage 2a: F[b] = W2cf * Hcat[b][0:512] + cf_b2  -> bf16 [128 x 1024]
    gemm_wmma<128, 128, 32, false><<<dim3(8, 1, 16), 256, 0, stream>>>(
        W2cf, 0LL, 512, Hcat, (long long)1536 * 1024, 1024,
        cf_b2, 0, nullptr, Fb, (long long)128 * 1024, 1024, 512, 0);
    // Stage 2b: p[b] = Wp[b] * Hcat[b][512:1536] + pbias[b] -> f32 [64 x 1024]
    gemm_wmma<64, 64, 16, false><<<dim3(16, 1, 16), 128, 0, stream>>>(
        Wp, (long long)64 * 1024, 1024, Hcat + (size_t)512 * 1024, (long long)1536 * 1024, 1024,
        pbias, 64, pbuf, nullptr, (long long)64 * 1024, 1024, 1024, 0);

    // Sinkhorn: 3 iterations of (u, v)
    for (int it = 0; it < 3; ++it) {
        sink_u<<<dim3(65, 16), 256, 0, stream>>>(pbuf, alpha, vbuf, ubuf);
        sink_v<<<dim3(16), 1024, 0, stream>>>(pbuf, alpha, ubuf, vbuf);
    }
    pexp_kernel<<<(16 * 1024 * 64) / 256, 256, 0, stream>>>(pbuf, ubuf, vbuf, Pt);

    // agg[b] = F[b] (128x1024) * Pt[b] (1024x64) -> f32 [128 x 64]
    gemm_wmma<64, 64, 16, false><<<dim3(1, 2, 16), 128, 0, stream>>>(
        Fb, (long long)128 * 1024, 1024, Pt, (long long)1024 * 64, 64,
        nullptr, 0, aggb, nullptr, (long long)128 * 64, 64, 1024, 0);

    finalize_kernel<<<16, 256, 0, stream>>>(tbuf, aggb, out);
}
